// MemoryCompressedAttention_68874095558665
// MI455X (gfx1250) — matrix operand order repaired and checked by reading, not compiled, not run on hardware
//
#include <hip/hip_runtime.h>
#include <cstdint>
#include <cstddef>

// ---------------------------------------------------------------------------
// Problem constants (MemoryCompressedAttention)
// ---------------------------------------------------------------------------
#define B_    4
#define S_    4096
#define D_    1024
#define H_    16
#define DK_   64
#define CR_   3
#define PAD_  2                       // CR - D % CR = 2
#define LP_   (S_ + PAD_)             // 4098 padded seq
#define SC_   (LP_ / CR_)             // 1366 compressed positions (3*1366 == 4098)
#define KCONV_ (CR_ * D_)             // 3072: conv == GEMM with K = 3*D

typedef __bf16 bf16_t;
typedef __attribute__((ext_vector_type(8)))  float  v8f;
typedef __attribute__((ext_vector_type(8)))  bf16_t v8bf;
typedef __attribute__((ext_vector_type(16))) bf16_t v16bf;

// ---------------------------------------------------------------------------
// Tensor Data Mover support (guarded; falls back to cooperative loads)
// ---------------------------------------------------------------------------
#if __has_builtin(__builtin_amdgcn_tensor_load_to_lds) && \
    __has_builtin(__builtin_amdgcn_s_wait_tensorcnt)
#define MCA_TDM 1
#else
#define MCA_TDM 0
#endif

typedef unsigned int tdm_v4u __attribute__((ext_vector_type(4)));
typedef int          tdm_v8i __attribute__((ext_vector_type(8)));
typedef int          tdm_v4i __attribute__((ext_vector_type(4)));

#if MCA_TDM
// 2D tile load: tile_w_elems x tile_rows (bf16 elements), global row stride
// row_stride_elems, zero-fill for rows >= tensor_rows (M-tail handling).
// LDS destination gets pad_amount DWORDs inserted every pad_interval DWORDs
// (codes per D# group1 spec) to reproduce the bank-conflict padding.
__device__ __forceinline__ void tdm_load_2d(const void* gptr, unsigned lds_off,
                                            unsigned tile_w_elems,
                                            unsigned tile_rows,
                                            unsigned long long row_stride_elems,
                                            unsigned tensor_rows,
                                            unsigned pad_interval_code,
                                            unsigned pad_amount_code) {
  const unsigned long long ga = (unsigned long long)(uintptr_t)gptr;
  tdm_v4u g0;
  g0[0] = 1u;                                    // count=1, user-mode load
  g0[1] = lds_off;                               // lds_addr (bytes)
  g0[2] = (unsigned)(ga & 0xFFFFFFFFull);        // global_addr[31:0]
  g0[3] = (unsigned)((ga >> 32) & 0x01FFFFFFull) | (2u << 30);  // [56:32]+type=2
  tdm_v8i g1;
  g1[0] = (int)((1u << 16)                        // data_size = 2 bytes
                | (1u << 20)                      // pad_enable
                | (pad_interval_code << 22) | (pad_amount_code << 25));
  g1[1] = (int)((tile_w_elems & 0xFFFFu) << 16);  // tensor_dim0[15:0]
  g1[2] = (int)((tile_w_elems >> 16) | ((tensor_rows & 0xFFFFu) << 16));
  g1[3] = (int)((tensor_rows >> 16) | ((tile_w_elems & 0xFFFFu) << 16)); // tile_dim0
  g1[4] = (int)(tile_rows & 0xFFFFu);             // tile_dim1 (tile_dim2 = 0)
  g1[5] = (int)(unsigned)(row_stride_elems & 0xFFFFFFFFull);  // dim0_stride lo
  g1[6] = (int)(unsigned)((row_stride_elems >> 32) & 0xFFFFull);
  g1[7] = 0;
  const tdm_v4i gz = {0, 0, 0, 0};
#if defined(__clang_major__) && (__clang_major__ >= 23)
  const tdm_v8i gz8 = {0, 0, 0, 0, 0, 0, 0, 0};
  __builtin_amdgcn_tensor_load_to_lds(g0, g1, gz, gz, gz8, 0);
#else
  __builtin_amdgcn_tensor_load_to_lds(g0, g1, gz, gz, 0);
#endif
}
#endif  // MCA_TDM

// ---------------------------------------------------------------------------
// Small helpers
// ---------------------------------------------------------------------------
__device__ __forceinline__ v8f vzero8f() {
  v8f r;
#pragma unroll
  for (int i = 0; i < 8; ++i) r[i] = 0.0f;
  return r;
}
__device__ __forceinline__ v8bf zero8bf() {
  v8bf r;
#pragma unroll
  for (int i = 0; i < 8; ++i) r[i] = (bf16_t)0.0f;
  return r;
}
__device__ __forceinline__ v8bf ldg8(const bf16_t* p) { return *(const v8bf*)p; }
__device__ __forceinline__ v16bf cat16(v8bf lo, v8bf hi) {
  v16bf r;
#pragma unroll
  for (int i = 0; i < 8; ++i) { r[i] = lo[i]; r[i + 8] = hi[i]; }
  return r;
}
__device__ __forceinline__ v8f wmma_bf16(v16bf a, v16bf b, v8f c) {
  return __builtin_amdgcn_wmma_f32_16x16x32_bf16(false, a, false, b, (short)0, c,
                                                 false, false);
}

// Max-reduce across the 16 lanes of a half-wave (butterfly). Prefers
// v_permlane16_b32 (pure VALU) over ds_bpermute-based __shfl_xor.
#if __has_builtin(__builtin_amdgcn_permlane16)
__device__ __forceinline__ float permx16(float v, unsigned s0, unsigned s1) {
  int u = __float_as_int(v);
  u = __builtin_amdgcn_permlane16(u, u, s0, s1, false, false);
  return __int_as_float(u);
}
__device__ __forceinline__ float redmax16(float v) {
  v = fmaxf(v, permx16(v, 0x67452301u, 0xEFCDAB89u));  // xor 1
  v = fmaxf(v, permx16(v, 0x54761032u, 0xDCFE98BAu));  // xor 2
  v = fmaxf(v, permx16(v, 0x32107654u, 0xBA98FEDCu));  // xor 4
  v = fmaxf(v, permx16(v, 0xFEDCBA98u, 0x76543210u));  // xor 8
  return v;
}
#else
__device__ __forceinline__ float redmax16(float v) {
  v = fmaxf(v, __shfl_xor(v, 1, 32));
  v = fmaxf(v, __shfl_xor(v, 2, 32));
  v = fmaxf(v, __shfl_xor(v, 4, 32));
  v = fmaxf(v, __shfl_xor(v, 8, 32));
  return v;
}
#endif

// ---------------------------------------------------------------------------
// Elementwise prep kernels
// ---------------------------------------------------------------------------
__global__ void k_f32_to_bf16(const float* __restrict__ s, bf16_t* __restrict__ d,
                              size_t n) {
  const size_t stride = (size_t)gridDim.x * blockDim.x;
  for (size_t i = (size_t)blockIdx.x * blockDim.x + threadIdx.x; i < n; i += stride)
    d[i] = (bf16_t)s[i];
}

__global__ void k_pad_convert(const float* __restrict__ s, bf16_t* __restrict__ d) {
  const size_t n = (size_t)B_ * LP_ * D_;
  const size_t stride = (size_t)gridDim.x * blockDim.x;
  for (size_t i = (size_t)blockIdx.x * blockDim.x + threadIdx.x; i < n; i += stride) {
    const size_t dd = i % D_;
    const size_t l  = (i / D_) % LP_;
    const size_t b  = i / ((size_t)D_ * LP_);
    float v = 0.0f;
    if (l >= PAD_) v = s[(b * S_ + (l - PAD_)) * D_ + dd];
    d[i] = (bf16_t)v;
  }
}

__global__ void k_rearrange_convw(const float* __restrict__ s, bf16_t* __restrict__ d) {
  const size_t n = (size_t)D_ * CR_ * D_;
  const size_t stride = (size_t)gridDim.x * blockDim.x;
  for (size_t i = (size_t)blockIdx.x * blockDim.x + threadIdx.x; i < n; i += stride) {
    const size_t o   = i / (CR_ * D_);
    const size_t rem = i % (CR_ * D_);
    const size_t r   = rem / D_;
    const size_t id  = rem % D_;
    d[i] = (bf16_t)s[(o * D_ + id) * CR_ + r];
  }
}

// ---------------------------------------------------------------------------
// Tiled WMMA GEMM:  out[M,N] = A[M,K] bf16 @ W[N,K]^T bf16 (+ bias)
// 256 threads / 8 waves; block tile 128x64; K-step 32; wave tile 32x32.
// TDM path: double-buffered LDS, waves 0/1 issue next-tile DMA before compute.
// ---------------------------------------------------------------------------
template <bool OUT_BF16>
__global__ __launch_bounds__(256) void gemm_bf16_wmma(
    const bf16_t* __restrict__ A, const bf16_t* __restrict__ W,
    const float* __restrict__ bias, void* __restrict__ out, int M, int N, int K) {
#if MCA_TDM
  __shared__ bf16_t As[2][128][40];   // 32 cols + 8 pad (== TDM pad 4 DW / 16 DW)
  __shared__ bf16_t Ws[2][64][40];
#else
  __shared__ bf16_t As[1][128][40];
  __shared__ bf16_t Ws[1][64][40];
#endif

  const int tid   = threadIdx.x;
  const int wid   = tid >> 5;
  const int lane  = tid & 31;
  const int wm    = wid & 3;
  const int wn    = wid >> 2;
  const int m0    = blockIdx.y * 128;
  const int n0    = blockIdx.x * 64;
  const int lm    = lane & 15;
  const int lhalf = lane >> 4;
  const int nIter = K >> 5;

  v8f acc[2][2];
#pragma unroll
  for (int mi = 0; mi < 2; ++mi)
#pragma unroll
    for (int ni = 0; ni < 2; ++ni) acc[mi][ni] = vzero8f();

#if MCA_TDM
  const unsigned arows = (unsigned)(M - m0);   // zero-fill past this row
  if (wid == 0)
    tdm_load_2d(A + (size_t)m0 * K, (unsigned)(uintptr_t)&As[0][0][0],
                32u, 128u, (unsigned long long)K, arows, 3u, 3u);
  else if (wid == 1)
    tdm_load_2d(W + (size_t)n0 * K, (unsigned)(uintptr_t)&Ws[0][0][0],
                32u, 64u, (unsigned long long)K, 64u, 3u, 3u);
  if (wid < 2) __builtin_amdgcn_s_wait_tensorcnt(0);
  __syncthreads();
#endif

  for (int it = 0; it < nIter; ++it) {
#if MCA_TDM
    const int buf = it & 1;
    if (it + 1 < nIter) {          // kick off next tile into the idle buffer
      const int k0n = (it + 1) << 5;
      if (wid == 0)
        tdm_load_2d(A + (size_t)m0 * K + k0n,
                    (unsigned)(uintptr_t)&As[buf ^ 1][0][0],
                    32u, 128u, (unsigned long long)K, arows, 3u, 3u);
      else if (wid == 1)
        tdm_load_2d(W + (size_t)n0 * K + k0n,
                    (unsigned)(uintptr_t)&Ws[buf ^ 1][0][0],
                    32u, 64u, (unsigned long long)K, 64u, 3u, 3u);
    }
#else
    const int buf = 0;
    const int k0  = it << 5;
    {
      const int ar = tid >> 1;            // 0..127
      const int ac = (tid & 1) * 16;      // 0/16
      const int grow = m0 + ar;
      if (grow < M) {
        const bf16_t* src = A + (size_t)grow * K + k0 + ac;
        *(v8bf*)&As[0][ar][ac]     = ldg8(src);
        *(v8bf*)&As[0][ar][ac + 8] = ldg8(src + 8);
        if (k0 + 32 < K) __builtin_prefetch(src + 32, 0, 0);
      } else {
        v8bf z = zero8bf();
        *(v8bf*)&As[0][ar][ac]     = z;
        *(v8bf*)&As[0][ar][ac + 8] = z;
      }
      const int wr = tid >> 2;            // 0..63
      const int wc = (tid & 3) * 8;
      *(v8bf*)&Ws[0][wr][wc] = ldg8(W + (size_t)(n0 + wr) * K + k0 + wc);
    }
    __syncthreads();
#endif

    const int hka = lhalf * 8;
    const int hkb = lhalf * 8;
    v16bf af[2], bfm[2];
#pragma unroll
    for (int mi = 0; mi < 2; ++mi) {
      const bf16_t* p = &As[buf][wm * 32 + mi * 16 + lm][hka];
      af[mi] = cat16(*(const v8bf*)p, *(const v8bf*)(p + 16));
    }
#pragma unroll
    for (int ni = 0; ni < 2; ++ni) {
      const bf16_t* p = &Ws[buf][wn * 32 + ni * 16 + lm][hkb];
      bfm[ni] = cat16(*(const v8bf*)p, *(const v8bf*)(p + 16));
    }
#pragma unroll
    for (int mi = 0; mi < 2; ++mi)
#pragma unroll
      for (int ni = 0; ni < 2; ++ni)
        acc[mi][ni] = wmma_bf16(af[mi], bfm[ni], acc[mi][ni]);

#if MCA_TDM
    if (it + 1 < nIter && wid < 2) __builtin_amdgcn_s_wait_tensorcnt(0);
#endif
    __syncthreads();
  }

#pragma unroll
  for (int ni = 0; ni < 2; ++ni) {
    const int col = n0 + wn * 32 + ni * 16 + lm;
    const float bv = bias ? bias[col] : 0.0f;
#pragma unroll
    for (int mi = 0; mi < 2; ++mi) {
#pragma unroll
      for (int r = 0; r < 8; ++r) {
        const int row = m0 + wm * 32 + mi * 16 + r + lhalf * 8;
        if (row < M) {
          const float v = acc[mi][ni][r] + bv;
          if (OUT_BF16)
            ((bf16_t*)out)[(size_t)row * N + col] = (bf16_t)v;
          else
            ((float*)out)[(size_t)row * N + col] = v;
        }
      }
    }
  }
}

// ---------------------------------------------------------------------------
// Flash attention over 1366 compressed keys.
// Grid (S/128, H, B); 8 waves; wave owns 16 query rows.
// Row sums of P come from an extra WMMA against an all-ones matrix (same
// recurrence as the output accumulator), so only the max needs a cross-lane
// reduction (v_permlane16). K tile staged by TDM when available.
// ---------------------------------------------------------------------------
__global__ __launch_bounds__(256) void mca_attention(
    const bf16_t* __restrict__ Q, const bf16_t* __restrict__ Kp,
    const bf16_t* __restrict__ Vp, bf16_t* __restrict__ O) {
  __shared__ bf16_t Ks[32][72];        // 32 keys x 64 dk (+8 pad == TDM pad)
  __shared__ bf16_t Vts[64][40];       // transposed: dk x keys (+8 pad)
  __shared__ bf16_t Ps[8][16][40];     // per-wave P transpose scratch

  const int tid   = threadIdx.x;
  const int wid   = tid >> 5;
  const int lane  = tid & 31;
  const int lm    = lane & 15;
  const int lhalf = lane >> 4;
  const int b     = blockIdx.z;
  const int h     = blockIdx.y;
  const int q0    = blockIdx.x * 128;
  const int qrow  = q0 + wid * 16;

  v16bf qf[2];
  {
    const bf16_t* qp = Q + (size_t)(b * S_ + qrow + lm) * D_ + h * DK_;
    const int hka = lhalf * 8;
#pragma unroll
    for (int ks = 0; ks < 2; ++ks)
      qf[ks] = cat16(ldg8(qp + ks * 32 + hka), ldg8(qp + ks * 32 + hka + 16));
  }

  v16bf onesf;
#pragma unroll
  for (int i = 0; i < 16; ++i) onesf[i] = (bf16_t)1.0f;

  float mst[8];
  v8f accO[4];
  v8f accL = vzero8f();                // running softmax denominator (P @ 1)
#pragma unroll
  for (int r = 0; r < 8; ++r) mst[r] = -1.0e30f;
#pragma unroll
  for (int t4 = 0; t4 < 4; ++t4) accO[t4] = vzero8f();

  const int NKB = (SC_ + 31) / 32;     // 43 key blocks
  for (int j = 0; j < NKB; ++j) {
    const int kb0 = j * 32;
    {
#if MCA_TDM
      if (wid == 0)
        tdm_load_2d(Kp + (size_t)(b * SC_ + kb0) * D_ + h * DK_,
                    (unsigned)(uintptr_t)&Ks[0][0], 64u, 32u,
                    (unsigned long long)D_, (unsigned)(SC_ - kb0), 4u, 3u);
#endif
      const int key = tid >> 3;
      const int c0  = (tid & 7) * 8;
      const int gk  = kb0 + key;
      v8bf vv = zero8bf();
#if !MCA_TDM
      v8bf kv = zero8bf();
#endif
      if (gk < SC_) {
        const size_t base = (size_t)(b * SC_ + gk) * D_ + h * DK_ + c0;
        vv = ldg8(Vp + base);
#if !MCA_TDM
        kv = ldg8(Kp + base);
#endif
      }
#if !MCA_TDM
      *(v8bf*)&Ks[key][c0] = kv;
#endif
#pragma unroll
      for (int e = 0; e < 8; ++e) Vts[c0 + e][key] = vv[e];
#if MCA_TDM
      if (wid == 0) __builtin_amdgcn_s_wait_tensorcnt(0);
#endif
    }
    __syncthreads();

    // scores S = (Q K^T) * 1/sqrt(DK)
    v8f sc2[2];
#pragma unroll
    for (int ni = 0; ni < 2; ++ni) {
      v8f s = vzero8f();
#pragma unroll
      for (int ks = 0; ks < 2; ++ks) {
        const bf16_t* p = &Ks[ni * 16 + lm][ks * 32 + lhalf * 8];
        v16bf kf = cat16(*(const v8bf*)p, *(const v8bf*)(p + 16));
        s = wmma_bf16(qf[ks], kf, s);
      }
      const int colg = kb0 + ni * 16 + lm;
#pragma unroll
      for (int r = 0; r < 8; ++r)
        s[r] = (colg < SC_) ? s[r] * 0.125f : -1.0e30f;
      sc2[ni] = s;
    }

    // online softmax: max via permlane butterfly; sum via ones-WMMA below
    float corr[8];
#pragma unroll
    for (int r = 0; r < 8; ++r) {
      const float rmx = redmax16(fmaxf(sc2[0][r], sc2[1][r]));
      const float mn  = fmaxf(mst[r], rmx);
      corr[r] = __expf(mst[r] - mn);
      mst[r]  = mn;
      sc2[0][r] = __expf(sc2[0][r] - mn);
      sc2[1][r] = __expf(sc2[1][r] - mn);
    }
#pragma unroll
    for (int t4 = 0; t4 < 4; ++t4)
#pragma unroll
      for (int r = 0; r < 8; ++r) accO[t4][r] *= corr[r];
#pragma unroll
    for (int r = 0; r < 8; ++r) accL[r] *= corr[r];

    // P: accumulator layout -> A-fragment layout via per-wave LDS scratch
#pragma unroll
    for (int ni = 0; ni < 2; ++ni)
#pragma unroll
      for (int r = 0; r < 8; ++r)
        Ps[wid][r + lhalf * 8][ni * 16 + lm] = (bf16_t)sc2[ni][r];
    __builtin_amdgcn_wave_barrier();   // keep intra-wave LDS RAW ordered

    v16bf pf;
    {
      const bf16_t* p = &Ps[wid][lm][lhalf * 8];
      pf = cat16(*(const v8bf*)p, *(const v8bf*)(p + 16));
    }
#pragma unroll
    for (int t4 = 0; t4 < 4; ++t4) {
      const bf16_t* vp2 = &Vts[t4 * 16 + lm][lhalf * 8];
      v16bf vf = cat16(*(const v8bf*)vp2, *(const v8bf*)(vp2 + 16));
      accO[t4] = wmma_bf16(pf, vf, accO[t4]);
    }
    accL = wmma_bf16(pf, onesf, accL);  // row sums, replicated across lanes
    __syncthreads();
  }

#pragma unroll
  for (int r = 0; r < 8; ++r) {
    const float inv = 1.0f / accL[r];
    const size_t row = (size_t)(b * S_ + qrow + r + lhalf * 8);
#pragma unroll
    for (int t4 = 0; t4 < 4; ++t4)
      O[row * D_ + h * DK_ + t4 * 16 + lm] = (bf16_t)(accO[t4][r] * inv);
  }
}

// ---------------------------------------------------------------------------
// Host-side orchestration (graph-capture safe: stream-ordered launches only)
// ---------------------------------------------------------------------------
extern "C" void kernel_launch(void* const* d_in, const int* in_sizes, int n_in,
                              void* d_out, int out_size, void* d_ws, size_t ws_size,
                              hipStream_t stream) {
  (void)in_sizes; (void)n_in; (void)out_size; (void)ws_size;
  const float* query  = (const float*)d_in[0];
  const float* key    = (const float*)d_in[1];
  const float* value  = (const float*)d_in[2];
  const float* Wq     = (const float*)d_in[3];
  const float* bq     = (const float*)d_in[4];
  const float* Wk     = (const float*)d_in[5];
  const float* bk     = (const float*)d_in[6];
  const float* Wv     = (const float*)d_in[7];
  const float* bv     = (const float*)d_in[8];
  const float* Wo     = (const float*)d_in[9];
  const float* bo     = (const float*)d_in[10];
  const float* conv_w = (const float*)d_in[11];
  const float* conv_b = (const float*)d_in[12];

  char* ws = (char*)d_ws;
  const size_t SZ_QBF = (size_t)B_ * S_ * D_ * 2;
  const size_t SZ_PAD = (size_t)B_ * LP_ * D_ * 2;
  const size_t SZ_W2  = (size_t)D_ * KCONV_ * 2;
  const size_t SZ_W   = (size_t)D_ * D_ * 2;
  const size_t SZ_C   = (size_t)B_ * SC_ * D_ * 2;

  size_t off = 0;
  bf16_t* Qbf  = (bf16_t*)(ws + off); off += SZ_QBF;   // later reused as AttnOut
  bf16_t* Kpad = (bf16_t*)(ws + off); off += SZ_PAD;   // later reused as Kproj
  bf16_t* Vpad = (bf16_t*)(ws + off); off += SZ_PAD;   // later reused as Vproj
  bf16_t* W2b  = (bf16_t*)(ws + off); off += SZ_W2;
  bf16_t* Wqb  = (bf16_t*)(ws + off); off += SZ_W;
  bf16_t* Wkb  = (bf16_t*)(ws + off); off += SZ_W;
  bf16_t* Wvb  = (bf16_t*)(ws + off); off += SZ_W;
  bf16_t* Wob  = (bf16_t*)(ws + off); off += SZ_W;
  bf16_t* Kc   = (bf16_t*)(ws + off); off += SZ_C;
  bf16_t* Vc   = (bf16_t*)(ws + off); off += SZ_C;
  bf16_t* Qp   = (bf16_t*)(ws + off); off += SZ_QBF;
  bf16_t* Kpj  = Kpad;  // Kpad dead after conv GEMMs
  bf16_t* Vpj  = Vpad;  // Vpad dead after conv GEMMs
  bf16_t* AO   = Qbf;   // Qbf dead after Q projection

  const dim3 blk(256);
  k_f32_to_bf16<<<2048, blk, 0, stream>>>(query, Qbf, (size_t)B_ * S_ * D_);
  k_pad_convert<<<2048, blk, 0, stream>>>(key, Kpad);
  k_pad_convert<<<2048, blk, 0, stream>>>(value, Vpad);
  k_rearrange_convw<<<1024, blk, 0, stream>>>(conv_w, W2b);
  k_f32_to_bf16<<<512, blk, 0, stream>>>(Wq, Wqb, (size_t)D_ * D_);
  k_f32_to_bf16<<<512, blk, 0, stream>>>(Wk, Wkb, (size_t)D_ * D_);
  k_f32_to_bf16<<<512, blk, 0, stream>>>(Wv, Wvb, (size_t)D_ * D_);
  k_f32_to_bf16<<<512, blk, 0, stream>>>(Wo, Wob, (size_t)D_ * D_);

  const int Mc = B_ * SC_;   // 5464 compressed rows
  const int Mq = B_ * S_;    // 16384 query rows
  const dim3 gc(D_ / 64, (Mc + 127) / 128);
  const dim3 gq(D_ / 64, Mq / 128);

  // Strided conv == GEMM over contiguous (B*Sc, 3*D) windows
  gemm_bf16_wmma<true><<<gc, blk, 0, stream>>>(Kpad, W2b, conv_b, Kc, Mc, D_, KCONV_);
  gemm_bf16_wmma<true><<<gc, blk, 0, stream>>>(Vpad, W2b, conv_b, Vc, Mc, D_, KCONV_);
  // Projections
  gemm_bf16_wmma<true><<<gq, blk, 0, stream>>>(Qbf, Wqb, bq, Qp, Mq, D_, D_);
  gemm_bf16_wmma<true><<<gc, blk, 0, stream>>>(Kc, Wkb, bk, Kpj, Mc, D_, D_);
  gemm_bf16_wmma<true><<<gc, blk, 0, stream>>>(Vc, Wvb, bv, Vpj, Mc, D_, D_);
  // Attention over compressed keys
  const dim3 ga(S_ / 128, H_, B_);
  mca_attention<<<ga, blk, 0, stream>>>(Qp, Kpj, Vpj, AO);
  // Output projection -> fp32 result
  gemm_bf16_wmma<false><<<gq, blk, 0, stream>>>(AO, Wob, bo, (float*)d_out, Mq, D_, D_);
}
